// RTNQuantizedLinear_81638738362813
// MI455X (gfx1250) — compile-verified, matrix operand order checked by reading, not run on hardware
//
#include <hip/hip_runtime.h>

// RTN 4-bit quantized linear as an f16-WMMA GEMM on gfx1250 (MI455X).
//   y[m,n] = sum_k x[m,k] * (q[n,k] - zp[n]) * scale[n] + bias[n]
// Dequant fused into the LDS tile loader as a single FMA per element:
//   w = q * scale + (-zp*scale)   (codes 0..15 exact in f16; |w| <~ 0.3)
// M=8192, N=11008, K=4096 -> compute-bound (AI ~1100 FLOP/B): feed the WMMA pipes.

typedef __attribute__((ext_vector_type(16))) _Float16 v16h;
typedef __attribute__((ext_vector_type(8)))  _Float16 v8h;
typedef __attribute__((ext_vector_type(4)))  _Float16 v4h;
typedef __attribute__((ext_vector_type(8)))  float    v8f;
typedef __attribute__((ext_vector_type(4)))  float    v4f;
typedef __attribute__((ext_vector_type(4)))  int      v4i;

#define IN_F   4096
#define OUT_F  11008
#define M_ROWS 8192          // 4 * 2048
#define BM 128
#define BN 128
#define BK 32
#define KT (IN_F / BK)       // 128 K-steps
#define LDS_STRIDE 40        // 32 halves + 8 pad halves (80B rows -> conflict-free b128 gathers)

__global__ __launch_bounds__(256)
void rtn_qlinear_wmma(const float* __restrict__ x,
                      const int*   __restrict__ qw,
                      const float* __restrict__ scale,
                      const float* __restrict__ zpt,
                      const float* __restrict__ bias,
                      float* __restrict__ out) {
  __shared__ _Float16 lA[2][BM * LDS_STRIDE];   // activations tile (f16)
  __shared__ _Float16 lW[2][BN * LDS_STRIDE];   // dequantized weights tile (f16)

  const int tid  = threadIdx.x;
  const int lane = tid & 31;
  const int half = lane >> 4;        // selects K-halves within a fragment
  const int lrow = lane & 15;        // A-row / B-col within a 16x16 fragment
  const int wv   = tid >> 5;         // wave id 0..7
  const int waveM = wv & 1;          // 2 waves along M (64 rows each)
  const int waveN = wv >> 1;         // 4 waves along N (32 cols each)
  const int mTile = blockIdx.y * BM;
  const int nTile = blockIdx.x * BN;

  // ---- per-thread loader setup: 4 slots of 4 elements cover a 128x32 tile ----
  const float* aP[4];
  const int*   wP[4];
  int   lOff[4];
  float sc[4], nzs[4];               // scale, -zp*scale (per output channel, K-invariant)
#pragma unroll
  for (int i = 0; i < 4; ++i) {
    const int s   = tid + i * 256;       // 0..1023
    const int row = s >> 3;              // 0..127
    const int cg  = s & 7;               // 4-element column group
    aP[i]   = x  + (size_t)(mTile + row) * IN_F + cg * 4;
    wP[i]   = qw + (size_t)(nTile + row) * IN_F + cg * 4;
    lOff[i] = row * LDS_STRIDE + cg * 4;
    const float s0 = scale[nTile + row];
    sc[i]  = s0;
    nzs[i] = -zpt[nTile + row] * s0;
  }

  v8f acc[4][2] = {};
  v4f aReg[4];
  v4i wReg[4];

  // ---- prologue: tile 0 -> regs -> LDS buf 0 ----
#pragma unroll
  for (int i = 0; i < 4; ++i) {
    aReg[i] = *(const v4f*)(aP[i]);
    wReg[i] = *(const v4i*)(wP[i]);
  }
#pragma unroll
  for (int i = 0; i < 4; ++i) {
    v4h ah, wh;
#pragma unroll
    for (int j = 0; j < 4; ++j) {
      ah[j] = (_Float16)aReg[i][j];
      wh[j] = (_Float16)__builtin_fmaf((float)wReg[i][j], sc[i], nzs[i]);  // v_pk_fma_f32
    }
    *(v4h*)&lA[0][lOff[i]] = ah;
    *(v4h*)&lW[0][lOff[i]] = wh;
  }

  int buf = 0;
  for (int kt = 0; kt < KT; ++kt) {
    __syncthreads();   // buf[next] stores from kt-1 visible; buf[cur] reads from kt-1 done

    // ---- issue global loads for tile kt+1 before the WMMA burst ----
    if (kt + 1 < KT) {
      const int kNext = (kt + 1) * BK;
#pragma unroll
      for (int i = 0; i < 4; ++i) {
        aReg[i] = *(const v4f*)(aP[i] + kNext);
        wReg[i] = *(const v4i*)(wP[i] + kNext);
      }
      if (kt + 8 < KT) {
        const int kPre = (kt + 8) * BK;
#pragma unroll
        for (int i = 0; i < 4; ++i) {
          __builtin_prefetch(aP[i] + kPre, 0, 1);   // global_prefetch_b8
          __builtin_prefetch(wP[i] + kPre, 0, 1);
        }
      }
    }

    // ---- compute from buf: 4x2 fragments, 8 v_wmma per K-step ----
    // 16-bit B fragment (32x16): lane -> col lrow; chunks at K = 8*half and 16+8*half
    v16h bfrag[2];
#pragma unroll
    for (int fn = 0; fn < 2; ++fn) {
      const _Float16* p = &lW[buf][(waveN * 32 + fn * 16 + lrow) * LDS_STRIDE];
      union { v16h v; v8h h[2]; } u;
      u.h[0] = *(const v8h*)(p + half * 8);
      u.h[1] = *(const v8h*)(p + 16 + half * 8);
      bfrag[fn] = u.v;
    }
#pragma unroll
    for (int fm = 0; fm < 4; ++fm) {
      // 16-bit A fragment (16x32): lane -> row lrow; same K-chunk pattern
      const _Float16* p = &lA[buf][(waveM * 64 + fm * 16 + lrow) * LDS_STRIDE];
      union { v16h v; v8h h[2]; } u;
      u.h[0] = *(const v8h*)(p + half * 8);
      u.h[1] = *(const v8h*)(p + 16 + half * 8);
      const v16h afrag = u.v;
      acc[fm][0] = __builtin_amdgcn_wmma_f32_16x16x32_f16(
          false, afrag, false, bfrag[0], (short)0, acc[fm][0], false, false);
      acc[fm][1] = __builtin_amdgcn_wmma_f32_16x16x32_f16(
          false, afrag, false, bfrag[1], (short)0, acc[fm][1], false, false);
    }

    // ---- convert + store tile kt+1 into the other buffer ----
    if (kt + 1 < KT) {
      const int nb = buf ^ 1;
#pragma unroll
      for (int i = 0; i < 4; ++i) {
        v4h ah, wh;
#pragma unroll
        for (int j = 0; j < 4; ++j) {
          ah[j] = (_Float16)aReg[i][j];
          wh[j] = (_Float16)__builtin_fmaf((float)wReg[i][j], sc[i], nzs[i]);
        }
        *(v4h*)&lA[nb][lOff[i]] = ah;
        *(v4h*)&lW[nb][lOff[i]] = wh;
      }
    }
    buf ^= 1;
  }

  // ---- epilogue: C layout is VGPR v -> M = v + 8*(lane/16), N = lane%16 ----
  // Output is written once, never re-read -> non-temporal stores keep L2 for x/weights.
#pragma unroll
  for (int fn = 0; fn < 2; ++fn) {
    const int col = nTile + waveN * 32 + fn * 16 + lrow;
    const float bv = bias[col];
#pragma unroll
    for (int fm = 0; fm < 4; ++fm) {
      const int r0 = mTile + waveM * 64 + fm * 16 + half * 8;
      float* op = out + (size_t)r0 * OUT_F + col;
#pragma unroll
      for (int v = 0; v < 8; ++v)
        __builtin_nontemporal_store(acc[fm][fn][v] + bv, op + (size_t)v * OUT_F);
    }
  }
}

extern "C" void kernel_launch(void* const* d_in, const int* in_sizes, int n_in,
                              void* d_out, int out_size, void* d_ws, size_t ws_size,
                              hipStream_t stream) {
  (void)in_sizes; (void)n_in; (void)out_size; (void)d_ws; (void)ws_size;
  const float* x     = (const float*)d_in[0];
  const int*   qw    = (const int*)  d_in[1];
  const float* scale = (const float*)d_in[2];
  const float* zp    = (const float*)d_in[3];
  const float* bias  = (const float*)d_in[4];
  float* out = (float*)d_out;

  dim3 grid(OUT_F / BN, M_ROWS / BM);   // 86 x 64 workgroups
  rtn_qlinear_wmma<<<grid, 256, 0, stream>>>(x, qw, scale, zp, bias, out);
}
